// LinAttention_15436112462235
// MI455X (gfx1250) — matrix operand order repaired and checked by reading, not compiled, not run on hardware
//
#include <hip/hip_runtime.h>

// LinAttention on MI455X (gfx1250), algebraically reassociated:
//   G[b]   = x[b]^T x[b]                      (WMMA bf16, f32 accum, split-K)
//   KtV    = Wk G Wv^T ; M[b] = sum_h Wq^T KtV (fp32 VALU, tiny)
//   out[b] = x[b] M[b]                        (WMMA bf16, f32 accum)
// ~4.3 GFLOP total, HBM-bound (~100 MB traffic).

typedef __attribute__((ext_vector_type(16))) __bf16 v16bf;
typedef __attribute__((ext_vector_type(8)))  __bf16 v8bf;
typedef __attribute__((ext_vector_type(8)))  float  v8f;

#define B_  8
#define N_  8192
#define D_  128
#define H_  8
#define KQ_ 64

__device__ __forceinline__ unsigned short f2bf(float f) {
  unsigned int u = __builtin_bit_cast(unsigned int, f);
  u += 0x7FFFu + ((u >> 16) & 1u);   // round-to-nearest-even
  return (unsigned short)(u >> 16);
}

// ---------------------------------------------------------------------------
// Kernel A: convert x (f32, [B][N][D]) -> xT (bf16, [B][D][N]) via LDS tile.
// Block handles 64 rows (n) x 128 cols (d). 1024 blocks x 256 threads.
// ---------------------------------------------------------------------------
__global__ void k_xT(const float* __restrict__ x, unsigned short* __restrict__ xT) {
  __shared__ unsigned short tile[128][66];   // pad 66 -> conflict-free
  const int t   = threadIdx.x;               // 0..255
  const int blk = blockIdx.x;                // 0..1023
  const int b   = blk >> 7;                  // /128
  const int n0  = (blk & 127) << 6;          // *64
  const float2* xp = (const float2*)(x + (size_t)(b * N_ + n0) * D_);
  // phase 1: coalesced load + convert, store transposed into LDS
  for (int i = 0; i < 16; ++i) {
    int p  = i * 256 + t;                    // 0..4095 float2's
    int nl = p >> 6;                         // local row
    int dp = p & 63;                         // d pair
    float2 v = xp[nl * 64 + dp];
    tile[2 * dp][nl]     = f2bf(v.x);
    tile[2 * dp + 1][nl] = f2bf(v.y);
  }
  __syncthreads();
  // phase 2: coalesced dword writes of xT rows
  unsigned int* o = (unsigned int*)xT;
  for (int j = 0; j < 16; ++j) {
    int q  = j * 256 + t;                    // 0..4095 dwords
    int d  = q >> 5;
    int cp = q & 31;                         // n pair within 64
    unsigned int lo = tile[d][2 * cp];
    unsigned int hi = tile[d][2 * cp + 1];
    o[(size_t)(b * D_ + d) * (N_ / 2) + (n0 >> 1) + cp] = lo | (hi << 16);
  }
}

// ---------------------------------------------------------------------------
// Kernel B: Gram partials Gp[seg][b][128][128] = xT_seg * xT_seg^T via WMMA.
// 4096 waves: (b, seg, ti, tj); each loops 1024 n-values (32 WMMAs).
// ---------------------------------------------------------------------------
__global__ void k_gram(const unsigned short* __restrict__ xT, float* __restrict__ Gp) {
  const int lane = threadIdx.x & 31;
  const int w    = blockIdx.x * 4 + (threadIdx.x >> 5);   // 0..4095
  const int b    = w >> 9;
  const int seg  = (w >> 6) & 7;
  const int ti   = (w >> 3) & 7;
  const int tj   = w & 7;
  const int half = lane >> 4;                // 0/1 (lane half)
  const int lm   = lane & 15;
  const unsigned short* arow = xT + (size_t)(b * D_ + ti * 16 + lm) * N_;
  const unsigned short* brow = xT + (size_t)(b * D_ + tj * 16 + lm) * N_;
  const int nbeg = seg * (N_ / 8), nend = nbeg + (N_ / 8);
  v8f acc = {};
  for (int n0 = nbeg; n0 < nend; n0 += 32) {
    union { v16bf v; v8bf h[2]; } A, Bm;
    // A (16x32, 16-bit): lane half 0 -> K {0..7,16..23}, half 1 -> {8..15,24..31}
    A.h[0]  = *(const v8bf*)(arow + n0 + half * 8);
    A.h[1]  = *(const v8bf*)(arow + n0 + 16 + half * 8);
    // B (32x16): lane half 0 -> K {0..7,16..23}, half 1 -> {8..15,24..31}
    Bm.h[0] = *(const v8bf*)(brow + n0 + half * 8);
    Bm.h[1] = *(const v8bf*)(brow + n0 + 16 + half * 8);
    acc = __builtin_amdgcn_wmma_f32_16x16x32_bf16(false, A.v, false, Bm.v,
                                                  (short)0, acc, false, false);
  }
  float* g = Gp + ((size_t)seg * B_ + b) * D_ * D_;
#pragma unroll
  for (int r = 0; r < 8; ++r) {
    int row = ti * 16 + r + half * 8;
    int col = tj * 16 + lm;
    g[row * D_ + col] = acc[r];
  }
}

// Reduce 8 split-K partials -> G[b][128][128] (deterministic, no atomics).
__global__ void k_gred(const float* __restrict__ Gp, float* __restrict__ G) {
  int f = blockIdx.x * 256 + threadIdx.x;    // 0..131071
  float s = 0.f;
#pragma unroll
  for (int seg = 0; seg < 8; ++seg) s += Gp[(size_t)seg * B_ * D_ * D_ + f];
  G[f] = s;
}

// ---------------------------------------------------------------------------
// Middle (tiny, fp32): T = Wk G ; KtV = T Wv^T ; MT[b][e][d] = sum_h Wq^T KtV
// ---------------------------------------------------------------------------
__global__ void k_T(const float* __restrict__ Wk, const float* __restrict__ G,
                    float* __restrict__ T) {
  int f  = blockIdx.x * 256 + threadIdx.x;   // 524288
  int d2 = f & 127;
  int i  = (f >> 7) & 63;
  int h  = (f >> 13) & 7;
  int b  = f >> 16;
  const float* wr = Wk + (size_t)(h * KQ_ + i) * D_;
  const float* gp = G + (size_t)b * D_ * D_ + d2;
  float acc = 0.f;
#pragma unroll 4
  for (int d = 0; d < D_; ++d) acc = fmaf(wr[d], gp[d * D_], acc);
  T[(size_t)((b * H_ + h) * KQ_ + i) * D_ + d2] = acc;
}

__global__ void k_KtV(const float* __restrict__ T, const float* __restrict__ Wv,
                      float* __restrict__ KtV) {
  int f = blockIdx.x * 256 + threadIdx.x;    // 524288
  int e = f & 127;
  int i = (f >> 7) & 63;
  int h = (f >> 13) & 7;
  int b = f >> 16;
  const float* tr = T + (size_t)((b * H_ + h) * KQ_ + i) * D_;
  const float* wv = Wv + (size_t)(h * D_ + e) * D_;
  float acc = 0.f;
#pragma unroll 4
  for (int d = 0; d < D_; ++d) acc = fmaf(tr[d], wv[d], acc);
  KtV[(size_t)((b * H_ + h) * KQ_ + i) * D_ + e] = acc;
}

__global__ void k_M(const float* __restrict__ Wq, const float* __restrict__ KtV,
                    unsigned short* __restrict__ MT) {
  int f = blockIdx.x * 256 + threadIdx.x;    // 131072
  int d = f & 127;
  int e = (f >> 7) & 127;
  int b = f >> 14;
  float acc = 0.f;
  for (int h = 0; h < H_; ++h) {
    const float* wq = Wq + (size_t)h * KQ_ * D_ + d;
    const float* kv = KtV + (size_t)((b * H_ + h) * KQ_) * D_ + e;
#pragma unroll 4
    for (int k = 0; k < KQ_; ++k) acc = fmaf(wq[k * D_], kv[k * D_], acc);
  }
  MT[(size_t)(b * D_ + e) * D_ + d] = f2bf(acc);   // store M transposed, bf16
}

// ---------------------------------------------------------------------------
// Kernel D: out[b] = x[b] @ M[b] via WMMA bf16 (x converted on the fly).
// 32768 waves: (b, nt in 512, et in 8); K = 128 -> 4 WMMAs.
// ---------------------------------------------------------------------------
__global__ void k_out(const float* __restrict__ x, const unsigned short* __restrict__ MT,
                      float* __restrict__ out) {
  const int lane = threadIdx.x & 31;
  const int w    = blockIdx.x * 4 + (threadIdx.x >> 5);   // 0..32767
  const int b    = w >> 12;
  const int nt   = (w >> 3) & 511;
  const int et   = w & 7;
  const int half = lane >> 4;
  const int lm   = lane & 15;
  const float*          xr = x + (size_t)(b * N_ + nt * 16 + lm) * D_;
  const unsigned short* mr = MT + (size_t)(b * D_ + et * 16 + lm) * D_;
  v8f acc = {};
  for (int d0 = 0; d0 < D_; d0 += 32) {
    union { v16bf v; v8bf h[2]; unsigned short u[16]; } A, Bm;
    float tmp[16];
    *(float4*)&tmp[0]  = *(const float4*)(xr + d0 + half * 8);
    *(float4*)&tmp[4]  = *(const float4*)(xr + d0 + half * 8 + 4);
    *(float4*)&tmp[8]  = *(const float4*)(xr + d0 + 16 + half * 8);
    *(float4*)&tmp[12] = *(const float4*)(xr + d0 + 16 + half * 8 + 4);
#pragma unroll
    for (int j = 0; j < 16; ++j) A.u[j] = f2bf(tmp[j]);
    Bm.h[0] = *(const v8bf*)(mr + d0 + half * 8);
    Bm.h[1] = *(const v8bf*)(mr + d0 + 16 + half * 8);
    acc = __builtin_amdgcn_wmma_f32_16x16x32_bf16(false, A.v, false, Bm.v,
                                                  (short)0, acc, false, false);
  }
#pragma unroll
  for (int r = 0; r < 8; ++r) {
    int n = nt * 16 + r + half * 8;
    out[(size_t)(b * N_ + n) * D_ + et * 16 + lm] = acc[r];
  }
}

// ---------------------------------------------------------------------------
extern "C" void kernel_launch(void* const* d_in, const int* in_sizes, int n_in,
                              void* d_out, int out_size, void* d_ws, size_t ws_size,
                              hipStream_t stream) {
  const float* x  = (const float*)d_in[0];
  const float* Wk = (const float*)d_in[1];
  const float* Wq = (const float*)d_in[2];
  const float* Wv = (const float*)d_in[3];
  float* out = (float*)d_out;

  char* ws = (char*)d_ws;
  // workspace layout (bytes)
  unsigned short* xT  = (unsigned short*)(ws + 0);          // 16,777,216
  float*          Gp  = (float*)(ws + 16777216);            //  4,194,304
  float*          G   = (float*)(ws + 20971520);            //    524,288
  float*          T   = (float*)(ws + 21495808);            //  2,097,152
  float*          KtV = (float*)(ws + 23592960);            //  2,097,152
  unsigned short* MT  = (unsigned short*)(ws + 25690112);   //    262,144
                                                            // total ~25.95 MB
  k_xT  <<<1024, 256, 0, stream>>>(x, xT);
  k_gram<<<1024, 128, 0, stream>>>(xT, Gp);
  k_gred<<< 512, 256, 0, stream>>>(Gp, G);
  k_T   <<<2048, 256, 0, stream>>>(Wk, G, T);
  k_KtV <<<2048, 256, 0, stream>>>(T, Wv, KtV);
  k_M   <<< 512, 256, 0, stream>>>(Wq, KtV, MT);
  k_out <<<8192, 128, 0, stream>>>(x, MT, out);
}
